// GraphNet_19344532701817
// MI455X (gfx1250) — matrix operand order repaired and checked by reading, not compiled, not run on hardware
//
#include <hip/hip_runtime.h>
#include <hip/hip_bf16.h>

// ---------------------------------------------------------------------------
// Problem constants (match reference)
// ---------------------------------------------------------------------------
#define D       256
#define N_DIAG  200
#define N_PRO   150
#define N_MED   400
#define NNODES  750           // N_DIAG + N_PRO + N_MED
#define NPAD    768           // padded to multiple of 16 for WMMA tiles
#define NH      3
#define NEG_SLOPE 0.2f

typedef __attribute__((ext_vector_type(16))) __bf16 v16bf;
typedef __attribute__((ext_vector_type(8)))  __bf16 v8bf;
typedef __attribute__((ext_vector_type(8)))  float  v8f;
typedef __attribute__((ext_vector_type(4)))  float  v4f;
typedef __attribute__((ext_vector_type(4)))  unsigned v4u;
typedef __attribute__((ext_vector_type(8)))  int    v8i;
typedef __attribute__((ext_vector_type(4)))  int    v4i;

// ---------------------------------------------------------------------------
// Utility kernels
// ---------------------------------------------------------------------------
__global__ void fill_zero_u32(unsigned* __restrict__ p, int n) {
  int t = blockIdx.x * blockDim.x + threadIdx.x;
  if (t < n) p[t] = 0u;
}

__global__ void gather_rows(const float* __restrict__ table,
                            const int* __restrict__ ids,
                            float* __restrict__ x) {
  int t = blockIdx.x * blockDim.x + threadIdx.x;   // over NPAD*D
  int i = t >> 8;
  int k = t & (D - 1);
  float v = 0.0f;
  if (i < NNODES) v = table[(size_t)ids[i] * D + k];
  x[t] = v;
}

// f32 -> bf16 row-major pack (n is a multiple of 256)
__global__ void pack_bf16(const float* __restrict__ src,
                          __bf16* __restrict__ dst, int n) {
  int t = blockIdx.x * blockDim.x + threadIdx.x;
  if (t < n) dst[t] = (__bf16)src[t];
}

// ---------------------------------------------------------------------------
// WMMA GEMM:  C[NPAD x 256] = A[NPAD x 256] @ W^T (+bias)
// A and W are pre-packed bf16 (row-major).  One wave per 16x16 C tile,
// K swept in steps of 32 with v_wmma_f32_16x16x32_bf16 (f32 accumulate,
// two independent accumulators to avoid the WMMA->WMMA hazard NOPs).
// Per block: the 64-column weight panel (64 x 256 bf16 = 32 KB) is staged
// into LDS by the Tensor Data Mover (tensor_load_to_lds, TENSORcnt), then
// all 4 waves consume it with ds_load_b128.
// grid = (NPAD/16, 4), block = 128 (4 waves).
// ---------------------------------------------------------------------------
__global__ __launch_bounds__(128)
void wmma_gemm_bf16(const __bf16* __restrict__ A,   // [NPAD][D] bf16
                    const __bf16* __restrict__ Wb,  // [D][D] bf16, row n holds W[n][k]
                    const float* __restrict__ bias, // nullable, length D
                    float* __restrict__ C) {
  __shared__ __bf16 bpanel[64 * D];                 // 32 KB, sole LDS object
  const int lane  = threadIdx.x & 31;
  const int wv    = threadIdx.x >> 5;
  const int l     = lane & 15;
  const int hi    = lane >> 4;                      // 0/1 (lane half)
  const int mtile = blockIdx.x << 4;
  const int nbase = blockIdx.y << 6;                // 64 N-columns per block
  const int ntile = nbase + (wv << 4);

  // ---- TDM: stage Wb rows [nbase, nbase+64) x K[0,256) into LDS ----------
  if (wv == 0) {
    // D# group 0: count=1 (valid), lds_addr, 57-bit global addr, type=2
    unsigned long long ga = (unsigned long long)(const void*)(Wb + (size_t)nbase * D);
    v4u g0;
    g0[0] = 1u;                                     // count=1, user mode
    g0[1] = 0u;                                     // lds_addr: bpanel @ offset 0
    g0[2] = (unsigned)(ga & 0xFFFFFFFFu);
    g0[3] = (unsigned)((ga >> 32) & 0x01FFFFFFu) | (2u << 30); // type=2 (image)
    // D# group 1: data_size=4B units; 2D tile 128 x 64, stride0 = 128 dwords
    v8i g1;
    g1[0] = (2 << 16);            // workgroup_mask=0, data_size=2 (4 bytes)
    g1[1] = (128 << 16);          // tensor_dim0 = 128 dword units (bits 79:48 lo)
    g1[2] = (64 << 16);           // tensor_dim0 hi=0 | tensor_dim1 = 64 rows
    g1[3] = (128 << 16);          // tensor_dim1 hi=0 | tile_dim0 = 128
    g1[4] = 64;                   // tile_dim1 = 64, tile_dim2 = 0
    g1[5] = 128;                  // tensor_dim0_stride = 128 dwords (row = 512 B)
    g1[6] = 0;
    g1[7] = 0;
    v4i z4 = {};
#if __clang_major__ >= 23
    v8i z8 = {};
    __builtin_amdgcn_tensor_load_to_lds(g0, g1, z4, z4, z8, 0);
#else
    __builtin_amdgcn_tensor_load_to_lds(g0, g1, z4, z4, 0);
#endif
    __builtin_amdgcn_s_wait_tensorcnt(0);
  }
  // The TDM writes bpanel behind the compiler's alias analysis.  Give the
  // compiler a visible (but dynamically never-executed: device VAs never
  // have the top nibble set) may-store with an opaque index so the LDS
  // reads below stay live and lower to ds_load_b128.
  if (__builtin_expect(((size_t)Wb >> 60) == 0xFull, 0))
    bpanel[(size_t)Wb & (64 * D - 1)] = (__bf16)0.0f;
  __syncthreads();   // LDS panel visible to all 4 waves

  const __bf16* arow = A + (size_t)(mtile + l) * D;
  const __bf16* brow = bpanel + (size_t)((wv << 4) + l) * D;   // column n of B

  v8f acc0 = {}, acc1 = {};
  #pragma unroll
  for (int kb = 0; kb < D; kb += 64) {
    // ---- K-step kb (accumulator 0) ----
    v8bf a0 = *(const v8bf*)(arow + kb + hi * 8);
    v8bf a1 = *(const v8bf*)(arow + kb + 16 + hi * 8);
    v8bf b0 = *(const v8bf*)(brow + kb + hi * 8);
    v8bf b1 = *(const v8bf*)(brow + kb + 16 + hi * 8);
    v16bf av = __builtin_shufflevector(a0, a1, 0, 1, 2, 3, 4, 5, 6, 7,
                                       8, 9, 10, 11, 12, 13, 14, 15);
    v16bf bv = __builtin_shufflevector(b0, b1, 0, 1, 2, 3, 4, 5, 6, 7,
                                       8, 9, 10, 11, 12, 13, 14, 15);
    acc0 = __builtin_amdgcn_wmma_f32_16x16x32_bf16(
        false, av, false, bv, (short)0, acc0, false, false);

    // ---- K-step kb+32 (accumulator 1, independent of acc0) ----
    v8bf a2 = *(const v8bf*)(arow + kb + 32 + hi * 8);
    v8bf a3 = *(const v8bf*)(arow + kb + 48 + hi * 8);
    v8bf b2 = *(const v8bf*)(brow + kb + 32 + hi * 8);
    v8bf b3 = *(const v8bf*)(brow + kb + 48 + hi * 8);
    v16bf aw = __builtin_shufflevector(a2, a3, 0, 1, 2, 3, 4, 5, 6, 7,
                                       8, 9, 10, 11, 12, 13, 14, 15);
    v16bf bw = __builtin_shufflevector(b2, b3, 0, 1, 2, 3, 4, 5, 6, 7,
                                       8, 9, 10, 11, 12, 13, 14, 15);
    acc1 = __builtin_amdgcn_wmma_f32_16x16x32_bf16(
        false, aw, false, bw, (short)0, acc1, false, false);
  }
  v8f acc = acc0 + acc1;

  // C layout: VGPR r -> M = mtile + hi*8 + r, N = ntile + (lane&15)
  const int   n  = ntile + l;
  const float bb = bias ? bias[n] : 0.0f;
  float* crow = C + (size_t)(mtile + hi * 8) * D + n;
  #pragma unroll
  for (int r = 0; r < 8; ++r) crow[r * D] = acc[r] + bb;
}

// ---------------------------------------------------------------------------
// Attention dot products: ssrc[h][i] = h[h][i] . att_src[h] (and dst)
// One wave per (h, i); 8 elements per lane + shuffle reduction.
// ---------------------------------------------------------------------------
__global__ __launch_bounds__(256)
void attn_dots(const float* __restrict__ hfeat,
               const float* __restrict__ att_src,
               const float* __restrict__ att_dst,
               float* __restrict__ ssrc,
               float* __restrict__ sdst) {
  int wid  = (blockIdx.x * blockDim.x + threadIdx.x) >> 5; // 0 .. NH*NPAD-1
  int lane = threadIdx.x & 31;
  int h = wid / NPAD;
  const float* hp = hfeat + (size_t)wid * D + lane * 8;
  const float* as = att_src + h * D + lane * 8;
  const float* ad = att_dst + h * D + lane * 8;
  v4f h0 = *(const v4f*)(hp);
  v4f h1 = *(const v4f*)(hp + 4);
  v4f s0 = *(const v4f*)(as);
  v4f s1 = *(const v4f*)(as + 4);
  v4f d0 = *(const v4f*)(ad);
  v4f d1 = *(const v4f*)(ad + 4);
  float s = 0.f, d = 0.f;
  #pragma unroll
  for (int j = 0; j < 4; ++j) {
    s += h0[j] * s0[j] + h1[j] * s1[j];
    d += h0[j] * d0[j] + h1[j] * d1[j];
  }
  #pragma unroll
  for (int m = 16; m >= 1; m >>= 1) {
    s += __shfl_xor(s, m, 32);
    d += __shfl_xor(d, m, 32);
  }
  if (lane == 0) { ssrc[wid] = s; sdst[wid] = d; }
}

// ---------------------------------------------------------------------------
// Edge softmax (3 passes over edge list, segment reduce over dst via atomics)
// ---------------------------------------------------------------------------
__device__ __forceinline__ unsigned f2key(float f) {
  unsigned u = __float_as_uint(f);
  return (u & 0x80000000u) ? ~u : (u | 0x80000000u);
}
__device__ __forceinline__ float key2f(unsigned k) {
  unsigned u = (k & 0x80000000u) ? (k & 0x7FFFFFFFu) : ~k;
  return __uint_as_float(u);
}

__global__ void edge_logits(const int* __restrict__ ei, int E,
                            const float* __restrict__ ssrc,
                            const float* __restrict__ sdst,
                            float* __restrict__ eatt,
                            unsigned* __restrict__ smax) {
  int t = blockIdx.x * blockDim.x + threadIdx.x;
  if (t >= NH * E) return;
  int h = t / E, e = t - h * E;
  int s = ei[e], d = ei[E + e];
  float v = ssrc[h * NPAD + s] + sdst[h * NPAD + d];
  v = (v > 0.f) ? v : NEG_SLOPE * v;                 // leaky_relu
  eatt[t] = v;
  atomicMax(&smax[h * NNODES + d], f2key(v));
}

__global__ void edge_expsum(const int* __restrict__ ei, int E,
                            const unsigned* __restrict__ smax,
                            float* __restrict__ eatt,
                            float* __restrict__ ssum) {
  int t = blockIdx.x * blockDim.x + threadIdx.x;
  if (t >= NH * E) return;
  int h = t / E, e = t - h * E;
  int d = ei[E + e];
  float a = __expf(eatt[t] - key2f(smax[h * NNODES + d]));
  eatt[t] = a;
  atomicAdd(&ssum[h * NNODES + d], a);
}

// One wave per (h, e): 256-wide message  msg = alpha * scale * h[src]
// accumulated into agg[h][dst][:] with f32 global atomics.
__global__ __launch_bounds__(256)
void edge_aggregate(const int* __restrict__ ei, int E,
                    const float* __restrict__ eatt,
                    const float* __restrict__ ssum,
                    const float* __restrict__ hfeat,
                    const float* __restrict__ edge_emb,
                    const int* __restrict__ ewi,
                    const int* __restrict__ ddi,   // null unless edge_dim==1
                    int edge_dim,
                    float* __restrict__ agg) {
  int wid  = (blockIdx.x * blockDim.x + threadIdx.x) >> 5;
  int lane = threadIdx.x & 31;
  if (wid >= NH * E) return;
  int h = wid / E, e = wid - h * E;
  int s = ei[e], d = ei[E + e];
  float alpha = eatt[wid] / ssum[h * NNODES + d];

  const float* hs = hfeat + (size_t)(h * NPAD + s) * D + lane * 8;
  float*       ag = agg   + (size_t)(h * NPAD + d) * D + lane * 8;
  v4f h0 = *(const v4f*)(hs);
  v4f h1 = *(const v4f*)(hs + 4);

  if (edge_dim == 1) {
    float sc = (edge_emb[ewi[e]] - (float)ddi[e]) * alpha;
    #pragma unroll
    for (int j = 0; j < 4; ++j) {
      atomicAdd(&ag[j],     sc * h0[j]);
      atomicAdd(&ag[4 + j], sc * h1[j]);
    }
  } else {
    const float* se = edge_emb + (size_t)ewi[e] * D + lane * 8;
    if (e + 1 < E)   // pull next edge's scale row toward L2/L0 (global_prefetch_b8)
      __builtin_prefetch(edge_emb + (size_t)ewi[e + 1] * D + lane * 8, 0, 1);
    v4f e0 = *(const v4f*)(se);
    v4f e1 = *(const v4f*)(se + 4);
    #pragma unroll
    for (int j = 0; j < 4; ++j) {
      atomicAdd(&ag[j],     alpha * e0[j] * h0[j]);
      atomicAdd(&ag[4 + j], alpha * e1[j] * h1[j]);
    }
  }
}

// ---------------------------------------------------------------------------
// Head combine:  out = relu( mean_h(agg_h + b_h) / denom );  store or accum
// ---------------------------------------------------------------------------
__global__ void combine_heads(const float* __restrict__ agg,
                              const float* __restrict__ head_b,
                              float inv_denom,
                              float* __restrict__ x1,
                              float* __restrict__ xacc,
                              int store_mode) {
  int t = blockIdx.x * blockDim.x + threadIdx.x;   // over NPAD*D
  int i = t >> 8;
  int k = t & (D - 1);
  float v = 0.0f;
  if (i < NNODES) {
    float s = 0.0f;
    #pragma unroll
    for (int h = 0; h < NH; ++h)
      s += agg[(size_t)(h * NPAD + i) * D + k] + head_b[h * D + k];
    v = s * (1.0f / 3.0f) * inv_denom;
    v = (v > 0.f) ? v : 0.f;
  }
  if (store_mode) { x1[t] = v; xacc[t] = v; }
  else if (i < NNODES) xacc[t] += v;
}

// Final segment sums over the three node ranges -> (3 x 256) output
__global__ void final_sums(const float* __restrict__ xacc,
                           float* __restrict__ out) {
  int k = threadIdx.x;   // 0..255
  float s1 = 0.f, s2 = 0.f, s3 = 0.f;
  for (int i = 0; i < N_DIAG; ++i)                    s1 += xacc[i * D + k];
  for (int i = N_DIAG; i < N_DIAG + N_PRO; ++i)       s2 += xacc[i * D + k];
  for (int i = N_DIAG + N_PRO; i < NNODES; ++i)       s3 += xacc[i * D + k];
  out[k]         = s1;
  out[D + k]     = s2;
  out[2 * D + k] = s3;
}

// ---------------------------------------------------------------------------
// Host side
// ---------------------------------------------------------------------------
struct Ws {
  float* x;       // NPAD*D
  float* xh;      // NPAD*D
  float* hfeat;   // NH*NPAD*D
  float* agg;     // NH*NPAD*D
  float* x1;      // NPAD*D
  float* xacc;    // NPAD*D
  float* ssrc;    // NH*NPAD
  float* sdst;    // NH*NPAD
  unsigned* smax; // NH*NNODES (padded)
  float* ssum;    // NH*NNODES (padded)
  float* eatt;    // NH*E_max
  __bf16* xb;     // NPAD*D   (bf16 of GEMM A input)
  __bf16* xhb;    // NPAD*D   (bf16 of xh)
  __bf16* wpack;  // 16 * D*D (bf16 weights: [lin, h0,h1,h2] x 4 relations)
};

static void gat_layer(const float* xin,
                      const __bf16* linWb, const float* lin_b,
                      const __bf16* headsWb,
                      const float* att_src, const float* att_dst,
                      const float* head_b,
                      const float* edge_emb, const int* ei, const int* ewi,
                      const int* ddi, int E, int edge_dim, float denom,
                      int store_mode, const Ws& w, hipStream_t stream) {
  // bf16 pack of the layer input, then lin projection (+optional bias)
  pack_bf16<<<(NPAD * D) / 256, 256, 0, stream>>>(xin, w.xb, NPAD * D);
  wmma_gemm_bf16<<<dim3(NPAD / 16, 4), 128, 0, stream>>>(w.xb, linWb, lin_b,
                                                         w.xh);
  // per-head projections from bf16(xh)
  pack_bf16<<<(NPAD * D) / 256, 256, 0, stream>>>(w.xh, w.xhb, NPAD * D);
  for (int h = 0; h < NH; ++h)
    wmma_gemm_bf16<<<dim3(NPAD / 16, 4), 128, 0, stream>>>(
        w.xhb, headsWb + (size_t)h * D * D, nullptr,
        w.hfeat + (size_t)h * NPAD * D);
  // attention scalar dots
  attn_dots<<<(NH * NPAD) / 8, 256, 0, stream>>>(w.hfeat, att_src, att_dst,
                                                 w.ssrc, w.sdst);
  // zero segment reduce buffers + message accumulators
  fill_zero_u32<<<(NH * NPAD * D + 255) / 256, 256, 0, stream>>>(
      (unsigned*)w.agg, NH * NPAD * D);
  fill_zero_u32<<<(NH * NNODES + 255) / 256, 256, 0, stream>>>(
      w.smax, NH * NNODES);
  fill_zero_u32<<<(NH * NNODES + 255) / 256, 256, 0, stream>>>(
      (unsigned*)w.ssum, NH * NNODES);

  int tot = NH * E;
  edge_logits<<<(tot + 255) / 256, 256, 0, stream>>>(ei, E, w.ssrc, w.sdst,
                                                     w.eatt, w.smax);
  edge_expsum<<<(tot + 255) / 256, 256, 0, stream>>>(ei, E, w.smax, w.eatt,
                                                     w.ssum);
  edge_aggregate<<<(tot + 7) / 8, 256, 0, stream>>>(
      ei, E, w.eatt, w.ssum, w.hfeat, edge_emb, ewi, ddi, edge_dim, w.agg);

  combine_heads<<<(NPAD * D) / 256, 256, 0, stream>>>(
      w.agg, head_b, 1.0f / denom, w.x1, w.xacc, store_mode);
}

extern "C" void kernel_launch(void* const* d_in, const int* in_sizes, int n_in,
                              void* d_out, int out_size, void* d_ws,
                              size_t ws_size, hipStream_t stream) {
  const float* node_table = (const float*)d_in[0];
  const int*   node_ids   = (const int*)d_in[1];
  // mm
  const float* mm_lin_W    = (const float*)d_in[2];
  const float* mm_lin_b    = (const float*)d_in[3];
  const float* mm_edge_emb = (const float*)d_in[4];
  const float* mm_heads_W  = (const float*)d_in[5];
  const float* mm_att_src  = (const float*)d_in[6];
  const float* mm_att_dst  = (const float*)d_in[7];
  const float* mm_head_b   = (const float*)d_in[8];
  const int*   mm_ei       = (const int*)d_in[9];
  const int*   mm_ewi      = (const int*)d_in[10];
  // dm
  const float* dm_lin_W    = (const float*)d_in[11];
  const float* dm_edge_emb = (const float*)d_in[12];
  const float* dm_heads_W  = (const float*)d_in[13];
  const float* dm_att_src  = (const float*)d_in[14];
  const float* dm_att_dst  = (const float*)d_in[15];
  const float* dm_head_b   = (const float*)d_in[16];
  const int*   dm_ei       = (const int*)d_in[17];
  const int*   dm_ewi      = (const int*)d_in[18];
  // pm
  const float* pm_lin_W    = (const float*)d_in[19];
  const float* pm_edge_emb = (const float*)d_in[20];
  const float* pm_heads_W  = (const float*)d_in[21];
  const float* pm_att_src  = (const float*)d_in[22];
  const float* pm_att_dst  = (const float*)d_in[23];
  const float* pm_head_b   = (const float*)d_in[24];
  const int*   pm_ei       = (const int*)d_in[25];
  const int*   pm_ewi      = (const int*)d_in[26];
  // dp
  const float* dp_lin_W    = (const float*)d_in[27];
  const float* dp_edge_emb = (const float*)d_in[28];
  const float* dp_heads_W  = (const float*)d_in[29];
  const float* dp_att_src  = (const float*)d_in[30];
  const float* dp_att_dst  = (const float*)d_in[31];
  const float* dp_head_b   = (const float*)d_in[32];
  const int*   dp_ei       = (const int*)d_in[33];
  const int*   dp_ewi      = (const int*)d_in[34];
  const int*   mm_ddi      = (const int*)d_in[35];

  const int E_mm = in_sizes[9] / 2;
  const int E_dm = in_sizes[17] / 2;
  const int E_pm = in_sizes[25] / 2;
  const int E_dp = in_sizes[33] / 2;

  // workspace carve-up (float region first, then bf16 region; 256-aligned)
  float* p = (float*)d_ws;
  Ws w;
  w.x     = p;                     p += NPAD * D;
  w.xh    = p;                     p += NPAD * D;
  w.hfeat = p;                     p += NH * NPAD * D;
  w.agg   = p;                     p += NH * NPAD * D;
  w.x1    = p;                     p += NPAD * D;
  w.xacc  = p;                     p += NPAD * D;
  w.ssrc  = p;                     p += NH * NPAD;
  w.sdst  = p;                     p += NH * NPAD;
  w.smax  = (unsigned*)p;          p += NH * NPAD;   // padded to NH*NPAD
  w.ssum  = p;                     p += NH * NPAD;
  w.eatt  = p;                     p += NH * 159744; // >= NH*E_mm, 256-aligned
  w.xb    = (__bf16*)p;            p += (NPAD * D) / 2;
  w.xhb   = (__bf16*)p;            p += (NPAD * D) / 2;
  w.wpack = (__bf16*)p;            /* 16 * D * D bf16 = 2 MB */

  // Pre-pack all 16 weight matrices to bf16: [lin, h0, h1, h2] per relation
  const float* lins[4]  = {mm_lin_W, dm_lin_W, pm_lin_W, dp_lin_W};
  const float* heads[4] = {mm_heads_W, dm_heads_W, pm_heads_W, dp_heads_W};
  for (int r = 0; r < 4; ++r) {
    pack_bf16<<<(D * D) / 256, 256, 0, stream>>>(
        lins[r], w.wpack + (size_t)(r * 4) * D * D, D * D);
    pack_bf16<<<(NH * D * D) / 256, 256, 0, stream>>>(
        heads[r], w.wpack + (size_t)(r * 4 + 1) * D * D, NH * D * D);
  }

  // Layer 0 (mm): gather -> lin(+bias) -> GAT, scalar edge scale, /N_MED
  gather_rows<<<(NPAD * D) / 256, 256, 0, stream>>>(node_table, node_ids, w.x);
  gat_layer(w.x, w.wpack + 0 * (size_t)D * D, mm_lin_b,
            w.wpack + 1 * (size_t)D * D, mm_att_src, mm_att_dst, mm_head_b,
            mm_edge_emb, mm_ei, mm_ewi, mm_ddi, E_mm,
            /*edge_dim=*/1, /*denom=*/(float)N_MED, /*store=*/1, w, stream);

  // Layers dm / pm / dp (all take x1), vector edge scale, accumulate into xacc
  gat_layer(w.x1, w.wpack + 4 * (size_t)D * D, nullptr,
            w.wpack + 5 * (size_t)D * D, dm_att_src, dm_att_dst, dm_head_b,
            dm_edge_emb, dm_ei, dm_ewi, nullptr, E_dm, D, (float)N_MED, 0, w,
            stream);
  gat_layer(w.x1, w.wpack + 8 * (size_t)D * D, nullptr,
            w.wpack + 9 * (size_t)D * D, pm_att_src, pm_att_dst, pm_head_b,
            pm_edge_emb, pm_ei, pm_ewi, nullptr, E_pm, D, (float)N_MED, 0, w,
            stream);
  gat_layer(w.x1, w.wpack + 12 * (size_t)D * D, nullptr,
            w.wpack + 13 * (size_t)D * D, dp_att_src, dp_att_dst, dp_head_b,
            dp_edge_emb, dp_ei, dp_ewi, nullptr, E_dp, D, (float)N_DIAG, 0, w,
            stream);

  final_sums<<<1, 256, 0, stream>>>(w.xacc, (float*)d_out);
}
